// IEGNNLayer_64433099374632
// MI455X (gfx1250) — matrix operand order repaired and checked by reading, not compiled, not run on hardware
//
#include <hip/hip_runtime.h>
#include <stdint.h>

#define NL 2048
#define NR 8192
#define DM 256
#define EL 65536
#define ER 262144
#define TE 32                      // edges per workgroup in k_edge

typedef _Float16 v8h  __attribute__((ext_vector_type(8)));
typedef _Float16 v16h __attribute__((ext_vector_type(16)));
typedef float    v8f  __attribute__((ext_vector_type(8)));
typedef unsigned int u32x4 __attribute__((ext_vector_type(4)));
typedef int      i32x4 __attribute__((ext_vector_type(4)));
typedef int      i32x8 __attribute__((ext_vector_type(8)));

#if __has_builtin(__builtin_amdgcn_tensor_load_to_lds) && __has_builtin(__builtin_amdgcn_s_wait_tensorcnt)
#define USE_TDM 1
#else
#define USE_TDM 0
#endif

// Assemble a 16-half WMMA operand from two contiguous 16B chunks.
__device__ __forceinline__ v16h ld2(const _Float16* p0, const _Float16* p1) {
  v8h lo = *(const v8h*)p0;
  v8h hi = *(const v8h*)p1;
  return __builtin_shufflevector(lo, hi, 0,1,2,3,4,5,6,7,8,9,10,11,12,13,14,15);
}

__device__ __forceinline__ v8f wmma16(v16h a, v16h b, v8f c) {
  // D = A(16x32 f16) * B(32x16 f16) + C(16x16 f32)
  return __builtin_amdgcn_wmma_f32_16x16x32_f16(false, a, false, b, (short)0, c, false, false);
}

__device__ __forceinline__ float siluf(float x) { return x / (1.0f + __expf(-x)); }

#if USE_TDM
// TDM load of a 2D tile: 32 rows x 256B (32 elems of 8B), contiguous rows,
// LDS padding 16B per 256B row -> matches a [32][136] f16 LDS array.
__device__ __forceinline__ void tdm_load_ktile(uint32_t lds_addr, const _Float16* gptr) {
  uint64_t ga = (uint64_t)(uintptr_t)gptr;
  u32x4 g0;
  g0.x = 1u;                                   // count=1 (valid), user mode, no gather
  g0.y = lds_addr;                             // LDS byte address (bits 63:32)
  g0.z = (uint32_t)ga;                         // global_addr[31:0]
  g0.w = (uint32_t)(ga >> 32) | (2u << 30);    // global_addr[56:32] | type=2 (image)
  // group1: wg_mask=0, data_size=3 (8B), pad_enable=1, pad_interval=5 (64 DW), pad_amount=3 (4 DW)
  uint32_t w0  = (3u << 16) | (1u << 20) | (5u << 22) | (3u << 25);
  uint32_t td0 = 0x40000000u;                  // huge tensor dims -> no OOB clamping
  uint32_t td1 = 0x40000000u;
  i32x8 g1;
  g1[0] = (int)w0;
  g1[1] = (int)((td0 & 0xFFFFu) << 16);                      // [63:48]=tensor_dim0 lo16
  g1[2] = (int)((td0 >> 16) | ((td1 & 0xFFFFu) << 16));      // dim0 hi16 | dim1 lo16
  g1[3] = (int)((td1 >> 16) | (32u << 16));                  // dim1 hi16 | tile_dim0=32
  g1[4] = (int)32u;                                          // tile_dim1=32, tile_dim2=0
  g1[5] = (int)32u;                                          // tensor_dim0_stride=32 elems
  g1[6] = 0;
  g1[7] = 0;
  i32x4 z4 = {0, 0, 0, 0};
#if defined(__clang_major__) && (__clang_major__ >= 23)
  i32x8 z8 = {0, 0, 0, 0, 0, 0, 0, 0};
  __builtin_amdgcn_tensor_load_to_lds(g0, g1, z4, z4, z8, 0);
#else
  __builtin_amdgcn_tensor_load_to_lds(g0, g1, z4, z4, 0);
#endif
}
#endif

// ---------------------------------------------------------------- utilities
__global__ void k_fill(float* __restrict__ p, int n) {
  int i = blockIdx.x * 256 + threadIdx.x;
  if (i < n) p[i] = 0.0f;
}

// f32 [rows,cols] -> f16 with output row stride ldo (halves)
__global__ void k_cvt16(const float* __restrict__ in, _Float16* __restrict__ out,
                        int rows, int cols, int ldo) {
  int i = blockIdx.x * 256 + threadIdx.x;
  if (i < rows * cols) {
    int r = i / cols, c = i % cols;
    out[(size_t)r * ldo + c] = (_Float16)in[i];
  }
}

// W f32 [k0..k0+K, N] row-major -> out f16 [N][K]  (B^T for WMMA)
__global__ void k_cvt_tr16(const float* __restrict__ W, int k0, int K, int N,
                           _Float16* __restrict__ out) {
  int i = blockIdx.x * 256 + threadIdx.x;
  if (i < K * N) {
    int n = i / K, k = i % K;
    out[i] = (_Float16)W[(size_t)(k0 + k) * N + n];
  }
}

// -------------------------------------------------- generic WMMA GEMM + bias + act
// One wave per 16x16 output tile. A: f16 [M,K] row stride lda. BT: f16 [N][K].
template<int ACT>
__global__ __launch_bounds__(32)
void k_gemm(const _Float16* __restrict__ A, int lda,
            const _Float16* __restrict__ BT,
            const float* __restrict__ bias,
            _Float16* __restrict__ o16, int ldo16,
            float* __restrict__ o32, int ldo32,
            int M, int N, int K) {
  int ntn = N >> 4;
  int mt = blockIdx.x / ntn;
  int nt = blockIdx.x - mt * ntn;
  int lane = threadIdx.x & 31;
  int r0    = lane & 15;
  int aoff  = (lane < 16) ? 0 : 8;    // A: K 0-7/16-23 vs 8-15/24-31
  int boff  = (lane < 16) ? 0 : 8;    // B: K 0-7/16-23 vs 8-15/24-31
  int mbase = (lane < 16) ? 0 : 8;    // C/D row split
  const _Float16* ap = A  + (size_t)(mt * 16 + r0) * lda + aoff;
  const _Float16* bp = BT + (size_t)(nt * 16 + r0) * K   + boff;
  v8f acc = {};
  for (int k = 0; k < K; k += 32) {
    if (k + 64 < K) {
      __builtin_prefetch(ap + k + 64, 0, 0);
      __builtin_prefetch(bp + k + 64, 0, 0);
    }
    v16h a = ld2(ap + k, ap + k + 16);
    v16h b = ld2(bp + k, bp + k + 16);
    acc = wmma16(a, b, acc);
  }
  int col = nt * 16 + r0;
  float bv = bias ? bias[col] : 0.0f;
  for (int r = 0; r < 8; r++) {
    float v = acc[r] + bv;
    if (ACT) v = siluf(v);
    size_t row = (size_t)(mt * 16 + mbase + r);
    if (o16) o16[row * ldo16 + col] = (_Float16)v;
    if (o32) o32[row * ldo32 + col] = v;
  }
}

// -------------------------------------------------- flash cross-attention
// 16 queries / workgroup (8 waves). 32-key chunks. Q,K: [*,128] f16. V: [Nk,256] f16.
// K tiles double-buffered through the Tensor Data Mover (pipelined TENSORcnt waits).
__global__ __launch_bounds__(256)
void k_flash(const _Float16* __restrict__ Q,
             const _Float16* __restrict__ K,
             const _Float16* __restrict__ V,
             _Float16* __restrict__ out, int ldo, int Nk) {
  __shared__ _Float16 sQ[16][136];
  __shared__ _Float16 sK[2][32][136];
  __shared__ _Float16 sVt[256][40];   // V^T tile: [col][key]
  __shared__ float    sS[16][32];
  __shared__ _Float16 sP[16][40];
  __shared__ float    s_m[16], s_l[16], s_scale[16];

  int tid = threadIdx.x;
  int lane = tid & 31, wave = tid >> 5;
  int q0 = blockIdx.x * 16;
  for (int i = tid; i < 16 * 16; i += 256) {         // Q tile: 16B vector copies
    int r = i >> 4, c8 = (i & 15) << 3;
    *(v8h*)&sQ[r][c8] = *(const v8h*)&Q[(size_t)(q0 + r) * 128 + c8];
  }
  if (tid < 16) { s_m[tid] = -3.0e38f; s_l[tid] = 0.0f; }
  v8f o0 = {}, o1 = {};
  int r0    = lane & 15;
  int aoff  = (lane < 16) ? 0 : 8;
  int boff  = (lane < 16) ? 0 : 8;
  int mbase = (lane < 16) ? 0 : 8;
#if USE_TDM
  if (wave == 0)                                     // prologue: async K tile 0
    tdm_load_ktile((uint32_t)(uintptr_t)&sK[0][0][0], K);
#endif
  __syncthreads();

  for (int kb = 0; kb < Nk; kb += 32) {
    int cur = (kb >> 5) & 1;
#if USE_TDM
    if (wave == 0) {
      if (kb + 32 < Nk) {                            // issue tile i+1, wait tile i
        tdm_load_ktile((uint32_t)(uintptr_t)&sK[cur ^ 1][0][0], K + (size_t)(kb + 32) * 128);
        __builtin_amdgcn_s_wait_tensorcnt(1);
      } else {
        __builtin_amdgcn_s_wait_tensorcnt(0);
      }
    }
#else
    for (int i = tid; i < 32 * 16; i += 256) {       // K tile: 16B vector copies
      int r = i >> 4, c8 = (i & 15) << 3;
      *(v8h*)&sK[cur][r][c8] = *(const v8h*)&K[(size_t)(kb + r) * 128 + c8];
    }
#endif
    for (int i = tid; i < 32 * 32; i += 256) {       // V tile: 16B loads, transposed stores
      int r = i >> 5, c8 = (i & 31) << 3;
      v8h v = *(const v8h*)&V[(size_t)(kb + r) * 256 + c8];
#pragma unroll
      for (int j = 0; j < 8; j++) sVt[c8 + j][r] = v[j];
    }
    if (kb + 32 < Nk)                                // warm next V tile in cache
      __builtin_prefetch(&V[(size_t)(kb + 32 + (tid & 31)) * 256 + ((tid >> 5) << 5)], 0, 0);
    __syncthreads();

    if (wave < 2) {                                  // S = Q K^T * scale
      v8f s = {};
      const _Float16* ap = &sQ[r0][aoff];
      const _Float16* bp = &sK[cur][wave * 16 + r0][boff];
      for (int k = 0; k < 128; k += 32) {
        v16h a = ld2(ap + k, ap + k + 16);
        v16h b = ld2(bp + k, bp + k + 16);
        s = wmma16(a, b, s);
      }
      for (int r = 0; r < 8; r++)
        sS[mbase + r][wave * 16 + r0] = s[r] * 0.08838834764831845f;  // 1/sqrt(128)
    }
    __syncthreads();

    if (tid < 16) {                                  // online softmax stats per row
      float mold = s_m[tid], mx = mold;
      for (int j = 0; j < 32; j++) mx = fmaxf(mx, sS[tid][j]);
      float sc = __expf(mold - mx);
      float l = s_l[tid] * sc;
      for (int j = 0; j < 32; j++) {
        float p = __expf(sS[tid][j] - mx);
        sP[tid][j] = (_Float16)p;
        l += p;
      }
      s_m[tid] = mx; s_l[tid] = l; s_scale[tid] = sc;
    }
    __syncthreads();

    for (int r = 0; r < 8; r++) {                    // rescale + O += P V
      float sc = s_scale[mbase + r];
      o0[r] *= sc; o1[r] *= sc;
    }
    v16h a = ld2(&sP[r0][aoff], &sP[r0][aoff + 16]);
    int c0 = wave * 32;
    v16h b0 = ld2(&sVt[c0 + r0][boff], &sVt[c0 + r0][boff + 16]);
    o0 = wmma16(a, b0, o0);
    v16h b1 = ld2(&sVt[c0 + 16 + r0][boff], &sVt[c0 + 16 + r0][boff + 16]);
    o1 = wmma16(a, b1, o1);
    __syncthreads();
  }

  for (int r = 0; r < 8; r++) {
    int row = mbase + r;
    float inv = 1.0f / s_l[row];
    size_t base = (size_t)(q0 + row) * ldo + wave * 32 + r0;
    out[base]      = (_Float16)(o0[r] * inv);
    out[base + 16] = (_Float16)(o1[r] * inv);
  }
}

// -------------------------------------------------- fused EGNN edge pipeline
// 32 edges / workgroup, two 16-edge M-tiles sharing every B (weight) fragment.
// hidden = silu(h[dst]W1a + h[src]W1b + d2*w1d2 + b1)
// m = silu(hidden W2 + b2) -> atomic scatter to magg; t = silu(m Cw1 + cb1)
// w = t . cw2 + cb2 -> atomic coord scatter.
__global__ __launch_bounds__(256)
void k_edge(const _Float16* __restrict__ h16, int ldh,
            const float* __restrict__ pos,
            const int* __restrict__ srcI, const int* __restrict__ dstI,
            const _Float16* __restrict__ W1aT, const _Float16* __restrict__ W1bT,
            const float* __restrict__ w1d2, const float* __restrict__ b1,
            const _Float16* __restrict__ W2T, const float* __restrict__ b2,
            const _Float16* __restrict__ Cw1T, const float* __restrict__ cb1,
            const float* __restrict__ cw2, const float* __restrict__ cb2,
            float* __restrict__ magg, float* __restrict__ coord_acc,
            float* __restrict__ cnt) {
  __shared__ _Float16 sh0[TE][264];
  __shared__ _Float16 sh1[TE][264];
  __shared__ float s_d2[TE];
  __shared__ float s_rel[TE][3];
  __shared__ int s_dst[TE], s_src[TE];

  int tid = threadIdx.x;
  int lane = tid & 31, wave = tid >> 5;
  int e0 = blockIdx.x * TE;
  if (tid < TE) {
    int e = e0 + tid;
    int s = srcI[e], d = dstI[e];
    s_src[tid] = s; s_dst[tid] = d;
    float rx = pos[d * 3 + 0] - pos[s * 3 + 0];
    float ry = pos[d * 3 + 1] - pos[s * 3 + 1];
    float rz = pos[d * 3 + 2] - pos[s * 3 + 2];
    s_rel[tid][0] = rx; s_rel[tid][1] = ry; s_rel[tid][2] = rz;
    s_d2[tid] = rx * rx + ry * ry + rz * rz;
    atomicAdd(&cnt[d], 1.0f);
  }
  __syncthreads();

  int r0    = lane & 15;
  int aoff  = (lane < 16) ? 0 : 8;
  int boff  = (lane < 16) ? 0 : 8;
  int mbase = (lane < 16) ? 0 : 8;
  int d0 = s_dst[r0],      d1 = s_dst[16 + r0];
  int sc0 = s_src[r0],     sc1 = s_src[16 + r0];

  // stage 1: two M-tiles share each B fragment
  for (int t = 0; t < 2; t++) {
    int nt = wave * 2 + t;
    v8f acc0 = {}, acc1 = {};
    const _Float16* bpa = W1aT + (size_t)(nt * 16 + r0) * 256 + boff;
    const _Float16* bpb = W1bT + (size_t)(nt * 16 + r0) * 256 + boff;
    const _Float16* pd0 = h16 + (size_t)d0 * ldh + aoff;
    const _Float16* pd1 = h16 + (size_t)d1 * ldh + aoff;
    const _Float16* ps0 = h16 + (size_t)sc0 * ldh + aoff;
    const _Float16* ps1 = h16 + (size_t)sc1 * ldh + aoff;
    for (int k = 0; k < 256; k += 32) {
      v16h b = ld2(bpa + k, bpa + k + 16);
      acc0 = wmma16(ld2(pd0 + k, pd0 + k + 16), b, acc0);
      acc1 = wmma16(ld2(pd1 + k, pd1 + k + 16), b, acc1);
    }
    for (int k = 0; k < 256; k += 32) {
      v16h b = ld2(bpb + k, bpb + k + 16);
      acc0 = wmma16(ld2(ps0 + k, ps0 + k + 16), b, acc0);
      acc1 = wmma16(ld2(ps1 + k, ps1 + k + 16), b, acc1);
    }
    int col = nt * 16 + r0;
    float bb = b1[col], wd = w1d2[col];
    for (int r = 0; r < 8; r++) {
      int row = mbase + r;
      sh0[row][col]      = (_Float16)siluf(acc0[r] + bb + s_d2[row] * wd);
      sh0[16 + row][col] = (_Float16)siluf(acc1[r] + bb + s_d2[16 + row] * wd);
    }
  }
  __syncthreads();

  // stage 2: m = silu(hidden W2 + b2); scatter into magg; keep f16 in sh1
  for (int t = 0; t < 2; t++) {
    int nt = wave * 2 + t;
    v8f acc0 = {}, acc1 = {};
    const _Float16* bp = W2T + (size_t)(nt * 16 + r0) * 256 + boff;
    const _Float16* p0 = &sh0[r0][aoff];
    const _Float16* p1 = &sh0[16 + r0][aoff];
    for (int k = 0; k < 256; k += 32) {
      v16h b = ld2(bp + k, bp + k + 16);
      acc0 = wmma16(ld2(p0 + k, p0 + k + 16), b, acc0);
      acc1 = wmma16(ld2(p1 + k, p1 + k + 16), b, acc1);
    }
    int col = nt * 16 + r0;
    float bb = b2[col];
    for (int r = 0; r < 8; r++) {
      int row = mbase + r;
      float v0 = siluf(acc0[r] + bb);
      float v1 = siluf(acc1[r] + bb);
      sh1[row][col]      = (_Float16)v0;
      sh1[16 + row][col] = (_Float16)v1;
      atomicAdd(&magg[(size_t)s_dst[row] * 256 + col], v0);
      atomicAdd(&magg[(size_t)s_dst[16 + row] * 256 + col], v1);
    }
  }
  __syncthreads();

  // stage 3: t = silu(m Cw1 + cb1) into sh0
  for (int t = 0; t < 2; t++) {
    int nt = wave * 2 + t;
    v8f acc0 = {}, acc1 = {};
    const _Float16* bp = Cw1T + (size_t)(nt * 16 + r0) * 256 + boff;
    const _Float16* p0 = &sh1[r0][aoff];
    const _Float16* p1 = &sh1[16 + r0][aoff];
    for (int k = 0; k < 256; k += 32) {
      v16h b = ld2(bp + k, bp + k + 16);
      acc0 = wmma16(ld2(p0 + k, p0 + k + 16), b, acc0);
      acc1 = wmma16(ld2(p1 + k, p1 + k + 16), b, acc1);
    }
    int col = nt * 16 + r0;
    float bb = cb1[col];
    for (int r = 0; r < 8; r++) {
      int row = mbase + r;
      sh0[row][col]      = (_Float16)siluf(acc0[r] + bb);
      sh0[16 + row][col] = (_Float16)siluf(acc1[r] + bb);
    }
  }
  __syncthreads();

  // stage 4: 4 edges per wave, lane-parallel dot + wave32 shfl reduction
  for (int e = wave * 4; e < wave * 4 + 4; e++) {
    float s = 0.0f;
    for (int k = lane; k < 256; k += 32) s += (float)sh0[e][k] * cw2[k];
    for (int offs = 16; offs > 0; offs >>= 1) s += __shfl_xor(s, offs, 32);
    if (lane == 0) {
      float w = s + cb2[0];
      int d = s_dst[e];
      atomicAdd(&coord_acc[(size_t)d * 3 + 0], s_rel[e][0] * w);
      atomicAdd(&coord_acc[(size_t)d * 3 + 1], s_rel[e][1] * w);
      atomicAdd(&coord_acc[(size_t)d * 3 + 2], s_rel[e][2] * w);
    }
  }
}

// -------------------------------------------------- node finalize
__global__ __launch_bounds__(256)
void k_finalize(const float* __restrict__ pos_in,
                const float* __restrict__ coord_acc,
                const float* __restrict__ cnt,
                const float* __restrict__ magg,
                _Float16* __restrict__ cat_right, int ldc,
                float* __restrict__ pos_out) {
  int n = blockIdx.x, t = threadIdx.x;
  float inv = 1.0f / fmaxf(cnt[n], 1.0f);
  cat_right[(size_t)n * ldc + t] = (_Float16)(magg[(size_t)n * 256 + t] * inv);
  if (t < 3)
    pos_out[(size_t)n * 3 + t] = pos_in[(size_t)n * 3 + t] + coord_acc[(size_t)n * 3 + t] * inv;
}

// ================================================================ host
extern "C" void kernel_launch(void* const* d_in, const int* in_sizes, int n_in,
                              void* d_out, int out_size, void* d_ws, size_t ws_size,
                              hipStream_t stream) {
  (void)in_sizes; (void)n_in; (void)out_size; (void)ws_size;

  const float* lig_x   = (const float*)d_in[0];
  const float* lig_pos = (const float*)d_in[1];
  const float* rec_x   = (const float*)d_in[2];
  const float* rec_pos = (const float*)d_in[3];
  const float* P[44];
  for (int i = 0; i < 44; i++) P[i] = (const float*)d_in[4 + i];
  const int* lig_ei = (const int*)d_in[48];
  const int* rec_ei = (const int*)d_in[49];

  enum { LQW = 0, LQB, RKW, RKB, RVW, RVB, RQW, RQB, LKW, LKB, LVW, LVB,
         LMW1, LMB1, LMW2, LMB2,
         LEW1, LEB1, LEW2, LEB2, LCW1, LCB1, LCW2, LCB2, LNW1, LNB1, LNW2, LNB2,
         RMW1, RMB1, RMW2, RMB2,
         REW1, REB1, REW2, REB2, RCW1, RCB1, RCW2, RCB2, RNW1, RNB1, RNW2, RNB2 };

  char* ws = (char*)d_ws;
  size_t off = 0;
  auto ha = [&](size_t n) -> _Float16* {
    off = (off + 255) & ~(size_t)255;
    _Float16* p = (_Float16*)(ws + off); off += n * sizeof(_Float16); return p;
  };
  auto fa = [&](size_t n) -> float* {
    off = (off + 255) & ~(size_t)255;
    float* p = (float*)(ws + off); off += n * sizeof(float); return p;
  };

  _Float16* catL  = ha((size_t)NL * 512);   // [lig_x16 | lig_ctx16]
  _Float16* catR  = ha((size_t)NR * 512);
  _Float16* catLN = ha((size_t)NL * 512);   // [u_lig16 | m_agg16]
  _Float16* catRN = ha((size_t)NR * 512);
  _Float16* Qlig = ha((size_t)NL * 128);
  _Float16* Krec = ha((size_t)NR * 128);
  _Float16* Vrec = ha((size_t)NR * 256);
  _Float16* Qrec = ha((size_t)NR * 128);
  _Float16* Klig = ha((size_t)NL * 128);
  _Float16* Vlig = ha((size_t)NL * 256);
  _Float16* tmp16 = ha((size_t)NR * 256);
  _Float16* WqligT = ha(128 * 256);
  _Float16* WkrecT = ha(128 * 256);
  _Float16* WvrecT = ha(256 * 256);
  _Float16* WqrecT = ha(128 * 256);
  _Float16* WkligT = ha(128 * 256);
  _Float16* WvligT = ha(256 * 256);
  _Float16* Lw1T = ha(256 * 512);
  _Float16* Lw2T = ha(256 * 256);
  _Float16* Rw1T = ha(256 * 512);
  _Float16* Rw2T = ha(256 * 256);
  _Float16* LE_W1aT = ha(256 * 256);
  _Float16* LE_W1bT = ha(256 * 256);
  _Float16* LE_W2T  = ha(256 * 256);
  _Float16* LE_Cw1T = ha(256 * 256);
  _Float16* LE_Nw1T = ha(256 * 512);
  _Float16* LE_Nw2T = ha(256 * 256);
  _Float16* RE_W1aT = ha(256 * 256);
  _Float16* RE_W1bT = ha(256 * 256);
  _Float16* RE_W2T  = ha(256 * 256);
  _Float16* RE_Cw1T = ha(256 * 256);
  _Float16* RE_Nw1T = ha(256 * 512);
  _Float16* RE_Nw2T = ha(256 * 256);
  float* magg = fa((size_t)NR * 256);
  float* cacc = fa((size_t)NR * 3);
  float* cnt  = fa((size_t)NR);

  float* out_ligx = (float*)d_out;
  float* out_ligp = out_ligx + (size_t)NL * 256;
  float* out_recx = out_ligp + (size_t)NL * 3;
  float* out_recp = out_recx + (size_t)NR * 256;

  auto cvtT = [&](const float* W, int k0, int K, int N, _Float16* o) {
    int n = K * N;
    k_cvt_tr16<<<(n + 255) / 256, 256, 0, stream>>>(W, k0, K, N, o);
  };
  auto gemm = [&](int act, const _Float16* A, int lda, const _Float16* BT, const float* bias,
                  _Float16* o16, int ldo16, float* o32, int ldo32, int M, int N, int K) {
    dim3 g((unsigned)((M / 16) * (N / 16)));
    if (act) k_gemm<1><<<g, 32, 0, stream>>>(A, lda, BT, bias, o16, ldo16, o32, ldo32, M, N, K);
    else     k_gemm<0><<<g, 32, 0, stream>>>(A, lda, BT, bias, o16, ldo16, o32, ldo32, M, N, K);
  };

  // ---- phase 0: f16 conversions / weight transposes
  k_cvt16<<<(NL * DM + 255) / 256, 256, 0, stream>>>(lig_x, catL, NL, DM, 512);
  k_cvt16<<<(NR * DM + 255) / 256, 256, 0, stream>>>(rec_x, catR, NR, DM, 512);
  cvtT(P[LQW], 0, 256, 128, WqligT);  cvtT(P[RKW], 0, 256, 128, WkrecT);
  cvtT(P[RVW], 0, 256, 256, WvrecT);  cvtT(P[RQW], 0, 256, 128, WqrecT);
  cvtT(P[LKW], 0, 256, 128, WkligT);  cvtT(P[LVW], 0, 256, 256, WvligT);
  cvtT(P[LMW1], 0, 512, 256, Lw1T);   cvtT(P[LMW2], 0, 256, 256, Lw2T);
  cvtT(P[RMW1], 0, 512, 256, Rw1T);   cvtT(P[RMW2], 0, 256, 256, Rw2T);
  cvtT(P[LEW1], 0, 256, 256, LE_W1aT);  cvtT(P[LEW1], 256, 256, 256, LE_W1bT);
  cvtT(P[LEW2], 0, 256, 256, LE_W2T);   cvtT(P[LCW1], 0, 256, 256, LE_Cw1T);
  cvtT(P[LNW1], 0, 512, 256, LE_Nw1T);  cvtT(P[LNW2], 0, 256, 256, LE_Nw2T);
  cvtT(P[REW1], 0, 256, 256, RE_W1aT);  cvtT(P[REW1], 256, 256, 256, RE_W1bT);
  cvtT(P[REW2], 0, 256, 256, RE_W2T);   cvtT(P[RCW1], 0, 256, 256, RE_Cw1T);
  cvtT(P[RNW1], 0, 512, 256, RE_Nw1T);  cvtT(P[RNW2], 0, 256, 256, RE_Nw2T);

  // ---- phase 1: QKV projections
  gemm(0, catL, 512, WqligT, P[LQB], Qlig, 128, nullptr, 0, NL, 128, 256);
  gemm(0, catR, 512, WkrecT, P[RKB], Krec, 128, nullptr, 0, NR, 128, 256);
  gemm(0, catR, 512, WvrecT, P[RVB], Vrec, 256, nullptr, 0, NR, 256, 256);
  gemm(0, catR, 512, WqrecT, P[RQB], Qrec, 128, nullptr, 0, NR, 128, 256);
  gemm(0, catL, 512, WkligT, P[LKB], Klig, 128, nullptr, 0, NL, 128, 256);
  gemm(0, catL, 512, WvligT, P[LVB], Vlig, 256, nullptr, 0, NL, 256, 256);

  // ---- phase 2: cross attention (ctx written into right half of cat buffers)
  k_flash<<<NL / 16, 256, 0, stream>>>(Qlig, Krec, Vrec, catL + 256, 512, NR);
  k_flash<<<NR / 16, 256, 0, stream>>>(Qrec, Klig, Vlig, catR + 256, 512, NL);

  // ---- phase 3: node update MLPs -> u (h for EGNN), into left half of catN
  gemm(1, catL, 512, Lw1T, P[LMB1], tmp16, 256, nullptr, 0, NL, 256, 512);
  gemm(0, tmp16, 256, Lw2T, P[LMB2], catLN, 512, nullptr, 0, NL, 256, 256);
  gemm(1, catR, 512, Rw1T, P[RMB1], tmp16, 256, nullptr, 0, NR, 256, 512);
  gemm(0, tmp16, 256, Rw2T, P[RMB2], catRN, 512, nullptr, 0, NR, 256, 256);

  // ---- phase 4a: ligand EGNN
  k_fill<<<(NL * 256 + 255) / 256, 256, 0, stream>>>(magg, NL * 256);
  k_fill<<<(NL * 3 + 255) / 256, 256, 0, stream>>>(cacc, NL * 3);
  k_fill<<<(NL + 255) / 256, 256, 0, stream>>>(cnt, NL);
  k_edge<<<EL / TE, 256, 0, stream>>>(catLN, 512, lig_pos, lig_ei, lig_ei + EL,
      LE_W1aT, LE_W1bT, P[LEW1] + (size_t)512 * 256, P[LEB1], LE_W2T, P[LEB2],
      LE_Cw1T, P[LCB1], P[LCW2], P[LCB2], magg, cacc, cnt);
  k_finalize<<<NL, 256, 0, stream>>>(lig_pos, cacc, cnt, magg, catLN + 256, 512, out_ligp);
  gemm(1, catLN, 512, LE_Nw1T, P[LNB1], tmp16, 256, nullptr, 0, NL, 256, 512);
  gemm(0, tmp16, 256, LE_Nw2T, P[LNB2], nullptr, 0, out_ligx, 256, NL, 256, 256);

  // ---- phase 4b: receptor EGNN
  k_fill<<<(NR * 256 + 255) / 256, 256, 0, stream>>>(magg, NR * 256);
  k_fill<<<(NR * 3 + 255) / 256, 256, 0, stream>>>(cacc, NR * 3);
  k_fill<<<(NR + 255) / 256, 256, 0, stream>>>(cnt, NR);
  k_edge<<<ER / TE, 256, 0, stream>>>(catRN, 512, rec_pos, rec_ei, rec_ei + ER,
      RE_W1aT, RE_W1bT, P[REW1] + (size_t)512 * 256, P[REB1], RE_W2T, P[REB2],
      RE_Cw1T, P[RCB1], P[RCW2], P[RCB2], magg, cacc, cnt);
  k_finalize<<<NR, 256, 0, stream>>>(rec_pos, cacc, cnt, magg, catRN + 256, 512, out_recp);
  gemm(1, catRN, 512, RE_Nw1T, P[RNB1], tmp16, 256, nullptr, 0, NR, 256, 512);
  gemm(0, tmp16, 256, RE_Nw2T, P[RNB2], nullptr, 0, out_recx, 256, NR, 256, 256);
}
